// TensHashCore_63256278335973
// MI455X (gfx1250) — 8-bit matrix operand order repaired and checked by reading, compiled, not run on hardware
//
#include <hip/hip_runtime.h>
#include <math.h>


typedef int   v16i __attribute__((ext_vector_type(16)));
typedef float v8f  __attribute__((ext_vector_type(8)));
typedef int   v4i_vs __attribute__((vector_size(16)));

#define HIDDEN   256
#define ROUNDS   64
#define M_TILE   128
#define THREADS  256

#define MAT_STRIDE 272
#define ST_STRIDE  272
#define MAT_BYTES  (HIDDEN * MAT_STRIDE)
#define ST_BYTES   (M_TILE * ST_STRIDE)
#define LDS_TOTAL  (2 * MAT_BYTES + ST_BYTES)

#ifndef __has_builtin
#define __has_builtin(x) 0
#endif
#if defined(__gfx1250__) && \
    __has_builtin(__builtin_amdgcn_global_load_async_to_lds_b128) && \
    __has_builtin(__builtin_amdgcn_s_wait_asynccnt)
#define USE_ASYNC_LDS 1
#else
#define USE_ASYNC_LDS 0
#endif

#if defined(__AMDGCN__) && __has_builtin(__builtin_amdgcn_fractf)
#define FRACTF(x) __builtin_amdgcn_fractf(x)
#else
#define FRACTF(x) ((x) - floorf(x))
#endif

#if USE_ASYNC_LDS
typedef __attribute__((address_space(1))) v4i_vs* gb128_p;
typedef __attribute__((address_space(3))) v4i_vs* lb128_p;
#endif

__device__ __forceinline__ unsigned int fp8h_of_ternary(float v) {
    return v > 0.5f ? 0x30u : (v < -0.5f ? 0xB0u : 0x00u);
}

__global__ void cvt_mats_fp8(const float* __restrict__ m,
                             unsigned int* __restrict__ out, int n4) {
    int i = blockIdx.x * blockDim.x + threadIdx.x;
    if (i >= n4) return;
    const float4 v = ((const float4*)m)[i];
    out[i] =  fp8h_of_ternary(v.x)
           | (fp8h_of_ternary(v.y) << 8)
           | (fp8h_of_ternary(v.z) << 16)
           | (fp8h_of_ternary(v.w) << 24);
}

__device__ __forceinline__ void copy_mat_row(const unsigned char* __restrict__ gsrc,
                                             unsigned char* ldst) {
#if USE_ASYNC_LDS
    gb128_p g = (gb128_p)gsrc;
    lb128_p l = (lb128_p)ldst;
    #pragma unroll
    for (int k = 0; k < 16; ++k) {
        __builtin_amdgcn_global_load_async_to_lds_b128(g + k, l + k, 0, 0);
    }
#else
    const uint4* s = (const uint4*)gsrc;
    uint4*       d = (uint4*)ldst;
    #pragma unroll
    for (int k = 0; k < 16; ++k) d[k] = s[k];
#endif
}

__global__ __launch_bounds__(THREADS, 1)
void tenshash_fp8(const float* __restrict__ bits,
                  const float* __restrict__ noise,
                  const unsigned char* __restrict__ mats8,
                  float* __restrict__ out) {
    extern __shared__ unsigned char smem[];
    unsigned char* sMat0 = smem;
    unsigned char* sMat1 = smem + MAT_BYTES;
    unsigned char* sSt   = smem + 2 * MAT_BYTES;

    const int t      = threadIdx.x;
    const int lane   = t & 31;
    const int wave   = t >> 5;
    const int laneLo = lane & 15;
    const int laneHi = lane >> 4;
    const int mrow0  = wave * 16;
    const long rowBase = (long)blockIdx.x * M_TILE;

    float nz[16][8];
    {
        const float* nrow0 = noise + (rowBase + mrow0 + laneHi * 8) * HIDDEN + laneLo;
        #pragma unroll
        for (int nt = 0; nt < 16; ++nt)
            #pragma unroll
            for (int d = 0; d < 8; ++d)
                nz[nt][d] = 0.5f * nrow0[d * HIDDEN + nt * 16];
    }

    #pragma unroll
    for (int g = 0; g < 32; ++g) {
        int e   = g * THREADS + t;
        int row = e >> 6;
        int cd  = e & 63;
        float4 v = ((const float4*)(bits + (rowBase + row) * HIDDEN))[cd];
        unsigned int b =  (v.x > 0.5f ? 0x38u : 0u)
                       | ((v.y > 0.5f ? 0x38u : 0u) << 8)
                       | ((v.z > 0.5f ? 0x38u : 0u) << 16)
                       | ((v.w > 0.5f ? 0x38u : 0u) << 24);
        *(unsigned int*)(sSt + row * ST_STRIDE + cd * 4) = b;
    }

    copy_mat_row(mats8 + (size_t)t * HIDDEN, sMat0 + t * MAT_STRIDE);

    for (int r = 0; r < ROUNDS; ++r) {
#if USE_ASYNC_LDS
        __builtin_amdgcn_s_wait_asynccnt(0);
#endif
        __syncthreads();

        unsigned char* cur = (r & 1) ? sMat1 : sMat0;
        if (r + 1 < ROUNDS) {
            unsigned char* nxt = (r & 1) ? sMat0 : sMat1;
            copy_mat_row(mats8 + (size_t)(r + 1) * (HIDDEN * HIDDEN) + (size_t)t * HIDDEN,
                         nxt + t * MAT_STRIDE);
        }

        v16i a0, a1;
        {
            const unsigned char* srow = sSt + (mrow0 + laneLo) * ST_STRIDE;
            #pragma unroll
            for (int p = 0; p < 8; ++p) {
                int k0 = (p >> 1) * 32 + laneHi * 16 + (p & 1) * 8;
                uint2 q0 = *(const uint2*)(srow + k0);
                uint2 q1 = *(const uint2*)(srow + k0 + 128);
                a0[2 * p]     = (int)q0.x;
                a0[2 * p + 1] = (int)q0.y;
                a1[2 * p]     = (int)q1.x;
                a1[2 * p + 1] = (int)q1.y;
            }
        }

        unsigned char* strow = sSt + (mrow0 + laneHi * 8) * ST_STRIDE + laneLo;

        #pragma unroll
        for (int nt = 0; nt < 16; ++nt) {
            v16i b0, b1;
            const unsigned char* brow =
                cur + (nt * 16 + laneLo) * MAT_STRIDE + laneHi * 16;
            #pragma unroll
            for (int g2 = 0; g2 < 4; ++g2) {
                uint4 u0 = *(const uint4*)(brow + g2 * 32);
                uint4 u1 = *(const uint4*)(brow + g2 * 32 + 128);
                b0[4 * g2]     = (int)u0.x;
                b0[4 * g2 + 1] = (int)u0.y;
                b0[4 * g2 + 2] = (int)u0.z;
                b0[4 * g2 + 3] = (int)u0.w;
                b1[4 * g2]     = (int)u1.x;
                b1[4 * g2 + 1] = (int)u1.y;
                b1[4 * g2 + 2] = (int)u1.z;
                b1[4 * g2 + 3] = (int)u1.w;
            }

            v8f acc;
            #pragma unroll
            for (int d = 0; d < 8; ++d) acc[d] = nz[nt][d];
            acc = __builtin_amdgcn_wmma_f32_16x16x128_fp8_fp8(
                      a0, b0, (short)0, acc, false, false);
            acc = __builtin_amdgcn_wmma_f32_16x16x128_fp8_fp8(
                      a1, b1, (short)0, acc, false, false);

            #pragma unroll
            for (int d = 0; d < 8; ++d) {
                float fr = FRACTF(acc[d]);
                strow[d * ST_STRIDE + nt * 16] =
                    (unsigned char)(fr >= 0.5f ? 0x38u : 0x00u);
            }
        }
    }

    __syncthreads();
    #pragma unroll
    for (int g = 0; g < 32; ++g) {
        int e   = g * THREADS + t;
        int row = e >> 6;
        int cd  = e & 63;
        unsigned int b = *(const unsigned int*)(sSt + row * ST_STRIDE + cd * 4);
        float4 v;
        v.x = (b & 0x000000FFu) ? 1.0f : 0.0f;
        v.y = (b & 0x0000FF00u) ? 1.0f : 0.0f;
        v.z = (b & 0x00FF0000u) ? 1.0f : 0.0f;
        v.w = (b & 0xFF000000u) ? 1.0f : 0.0f;
        ((float4*)(out + (rowBase + row) * HIDDEN))[cd] = v;
    }
}

extern "C" void kernel_launch(void* const* d_in, const int* in_sizes, int n_in,
                              void* d_out, int out_size, void* d_ws, size_t ws_size,
                              hipStream_t stream) {
    const float* bits  = (const float*)d_in[0];
    const float* noise = (const float*)d_in[1];
    const float* mats  = (const float*)d_in[2];
    float* out = (float*)d_out;
    unsigned char* mats8 = (unsigned char*)d_ws;

    int n4 = (ROUNDS * HIDDEN * HIDDEN) / 4;
    hipLaunchKernelGGL(cvt_mats_fp8, dim3((n4 + 255) / 256), dim3(256), 0, stream,
                       mats, (unsigned int*)mats8, n4);

    int rows   = in_sizes[0] / HIDDEN;
    int blocks = rows / M_TILE;
    hipLaunchKernelGGL(tenshash_fp8, dim3(blocks), dim3(THREADS), LDS_TOTAL, stream,
                       bits, noise, mats8, out);
}
